// QALoRALinear_67843303408244
// MI455X (gfx1250) — 8-bit matrix operand order repaired and checked by reading, compiled, not run on hardware
//
#include <hip/hip_runtime.h>


typedef __attribute__((ext_vector_type(8))) int   v8i;
typedef __attribute__((ext_vector_type(4))) int   v4i;
typedef __attribute__((ext_vector_type(4))) float v4f;

#define IN_F   4096
#define OUT_F  4096
#define RANK   16
#define M_TOT  8192
#define LORA_SCALING 0.01f
#define EPS_SC 1e-8f
#define QMINF  (-8.0f)
#define QMAXF  (7.0f)

__global__ __launch_bounds__(256) void qmerge_weight_kernel(
    const float* __restrict__ W,
    const float* __restrict__ lA,
    const float* __restrict__ lB,
    const float* __restrict__ wsc,
    signed char* __restrict__ qw)
{
    const int o = blockIdx.x;
    float b[RANK];
#pragma unroll
    for (int r = 0; r < RANK; ++r) b[r] = LORA_SCALING * lB[o * RANK + r];

    const float s   = fmaxf(fabsf(wsc[o]), EPS_SC);
    const float inv = 1.0f / s;

    for (int c = threadIdx.x; c < IN_F; c += blockDim.x) {
        float acc = W[(size_t)o * IN_F + c];
#pragma unroll
        for (int r = 0; r < RANK; ++r)
            acc = fmaf(b[r], lA[r * IN_F + c], acc);
        float q = fminf(fmaxf(rintf(acc * inv), QMINF), QMAXF);
        qw[(size_t)o * IN_F + c] = (signed char)(int)q;
    }
}

__global__ __launch_bounds__(256) void qx_kernel(
    const float* __restrict__ x,
    const float* __restrict__ asc_p,
    int*         __restrict__ qx4,
    int n4)
{
    const float asc = fmaxf(fabsf(asc_p[0]), EPS_SC);
    const float inv = 1.0f / asc;
    int i = blockIdx.x * blockDim.x + threadIdx.x;
    if (i < n4) {
        v4f v = ((const v4f*)x)[i];
        int c0 = (int)fminf(fmaxf(rintf(v.x * inv), QMINF), QMAXF);
        int c1 = (int)fminf(fmaxf(rintf(v.y * inv), QMINF), QMAXF);
        int c2 = (int)fminf(fmaxf(rintf(v.z * inv), QMINF), QMAXF);
        int c3 = (int)fminf(fmaxf(rintf(v.w * inv), QMINF), QMAXF);
        qx4[i] = (c0 & 0xff) | ((c1 & 0xff) << 8) | ((c2 & 0xff) << 16) | (c3 << 24);
    }
}

__device__ __forceinline__ v8i load_a_frag(const signed char* bk, int voff) {
    union { v8i v; long l[4]; } u;
    u.l[0] = *(const long*)(bk + voff);
    u.l[1] = *(const long*)(bk + voff + 8);
    u.l[2] = *(const long*)(bk + voff + 32);
    u.l[3] = *(const long*)(bk + voff + 40);
    return u.v;
}

__device__ __forceinline__ v8i load_b_frag(const signed char* bk, int voff) {
    union { v8i v; v4i h[2]; } u;
    u.h[0] = *(const v4i*)(bk + voff);
    u.h[1] = *(const v4i*)(bk + voff + 32);
    return u.v;
}

__device__ __forceinline__ void load_set(const signed char* __restrict__ qx_k,
                                         const signed char* __restrict__ qw_k,
                                         const int aoff[4], const int boff[4],
                                         v8i a[4], v8i b[4]) {
#pragma unroll
    for (int i = 0; i < 4; ++i) a[i] = load_a_frag(qx_k, aoff[i]);
#pragma unroll
    for (int j = 0; j < 4; ++j) b[j] = load_b_frag(qw_k, boff[j]);
}

__device__ __forceinline__ void wmma_step(v8i acc[4][4], const v8i a[4], const v8i b[4]) {
#pragma unroll
    for (int i = 0; i < 4; ++i)
#pragma unroll
        for (int j = 0; j < 4; ++j)
            acc[i][j] = __builtin_amdgcn_wmma_i32_16x16x64_iu8(
                 true, a[i],  true, b[j],
                acc[i][j],  false,  false);
}

#define BM 128
#define BN 256

__global__ __launch_bounds__(256, 1) void gemm_iu8_wmma_kernel(
    const signed char* __restrict__ qx,
    const signed char* __restrict__ qw,
    const float* __restrict__ wsc,
    const float* __restrict__ asc_p,
    const float* __restrict__ bias,
    float* __restrict__ out)
{
    const int lane = threadIdx.x & 31;
    const int wid  = threadIdx.x >> 5;
    const int wm   = wid & 1;
    const int wn   = wid >> 1;
    const int mbase = blockIdx.y * BM + wm * 64;
    const int nbase = blockIdx.x * BN + wn * 64;
    const int l15 = lane & 15;
    const int lhi = lane >> 4;

    int aoff[4], boff[4];
#pragma unroll
    for (int i = 0; i < 4; ++i) {
        aoff[i] = (mbase + i * 16 + l15) * IN_F + lhi * 16;
        boff[i] = (nbase + i * 16 + l15) * IN_F + lhi * 16;
    }

    v8i acc[4][4];
#pragma unroll
    for (int i = 0; i < 4; ++i)
#pragma unroll
        for (int j = 0; j < 4; ++j)
            acc[i][j] = (v8i)0;

    v8i a0[4], b0[4], a1[4], b1[4];

    load_set(qx, qw, aoff, boff, a0, b0);

#pragma unroll 1
    for (int k0 = 0; k0 < IN_F - 128; k0 += 128) {
        load_set(qx + k0 + 64, qw + k0 + 64, aoff, boff, a1, b1);
#pragma unroll
        for (int i = 0; i < 4; ++i) {
            __builtin_prefetch(qx + k0 + 256 + aoff[i], 0, 1);
            __builtin_prefetch(qw + k0 + 256 + boff[i], 0, 1);
        }
        wmma_step(acc, a0, b0);
        load_set(qx + k0 + 128, qw + k0 + 128, aoff, boff, a0, b0);
        wmma_step(acc, a1, b1);
    }
    load_set(qx + IN_F - 64, qw + IN_F - 64, aoff, boff, a1, b1);
    wmma_step(acc, a0, b0);
    wmma_step(acc, a1, b1);

    const float asc = fmaxf(fabsf(asc_p[0]), EPS_SC);
#pragma unroll
    for (int j = 0; j < 4; ++j) {
        const int ncol = nbase + j * 16 + l15;
        const float sf = fmaxf(fabsf(wsc[ncol]), EPS_SC) * asc;
        const float bs = bias[ncol];
#pragma unroll
        for (int i = 0; i < 4; ++i) {
#pragma unroll
            for (int v = 0; v < 8; ++v) {
                const int row = mbase + i * 16 + v + lhi * 8;
                out[(size_t)row * OUT_F + ncol] = (float)acc[i][j][v] * sf + bs;
            }
        }
    }
}

extern "C" void kernel_launch(void* const* d_in, const int* in_sizes, int n_in,
                              void* d_out, int out_size, void* d_ws, size_t ws_size,
                              hipStream_t stream) {
    const float* x    = (const float*)d_in[0];
    const float* W    = (const float*)d_in[1];
    const float* lA   = (const float*)d_in[2];
    const float* lB   = (const float*)d_in[3];
    const float* wsc  = (const float*)d_in[4];
    const float* asc  = (const float*)d_in[5];
    const float* bias = (const float*)d_in[6];

    signed char* qw = (signed char*)d_ws;
    signed char* qx = (signed char*)d_ws + (size_t)OUT_F * IN_F;
    float* out = (float*)d_out;

    qmerge_weight_kernel<<<OUT_F, 256, 0, stream>>>(W, lA, lB, wsc, qw);

    const int n4 = (M_TOT * IN_F) / 4;
    qx_kernel<<<(n4 + 255) / 256, 256, 0, stream>>>(x, asc, (int*)qx, n4);

    dim3 grid(OUT_F / BN, M_TOT / BM);
    gemm_iu8_wmma_kernel<<<grid, 256, 0, stream>>>(qx, qw, wsc, asc, bias, out);
}
